// BinaryDense_10136122818965
// MI455X (gfx1250) — matrix operand order repaired and checked by reading, not compiled, not run on hardware
//
#include <hip/hip_runtime.h>
#include <cstdint>

// ---------------------------------------------------------------------------
// BinaryDense forward on MI455X (gfx1250):
//   C = X @ (w_raw > 0)            X: 4096x4096 fp32, out fp32
// Binary weights are exact in bf16 (0/1). Split X into two bf16 terms
// (hi + lo, RNE); both products against {0,1} are exact, accumulated in f32
// WMMA -> fp32-class accuracy at 2x bf16-WMMA cost (274 GFLOP of WMMA work).
// All bf16 operands (32+32+32 MB) fit in the 192 MB L2, so the GEMM is
// matrix-core bound. W is pre-transposed (N x K) once so the hot loop is
// purely b128 global loads -> b128 LDS stores -> b128 fragment loads -> WMMA.
// ---------------------------------------------------------------------------

typedef __attribute__((ext_vector_type(16))) __bf16    v16bf;
typedef __attribute__((ext_vector_type(8)))  float     v8f;
typedef __attribute__((ext_vector_type(4)))  uint32_t  u32x4;
typedef __attribute__((ext_vector_type(4)))  float     f32x4;

static constexpr int MDIM = 4096;   // batch rows
static constexpr int NDIM = 4096;   // feature cols
static constexpr int KDIM = 4096;   // d_in

static constexpr int BM = 128, BN = 128, BK = 32;
static constexpr int LDSA = 40;     // padded LDS row stride (halfs): 80B, 16B-aligned
static constexpr int LDSB = 40;

// ---- fp32 <-> bf16 bit helpers (round-to-nearest-even) ----
__device__ __forceinline__ uint16_t f32_to_bf16(float f) {
    union { float f; uint32_t u; } c; c.f = f;
    uint32_t u = c.u;
    uint32_t r = u + 0x7FFFu + ((u >> 16) & 1u);
    return (uint16_t)(r >> 16);
}
__device__ __forceinline__ float bf16_to_f32(uint16_t h) {
    union { uint32_t u; float f; } c; c.u = ((uint32_t)h) << 16;
    return c.f;
}

// Split X into hi/lo bf16 terms: x = hi + lo (lo absorbs the hi rounding error).
__global__ void __launch_bounds__(256)
cvt_x_split(const float* __restrict__ x,
            uint16_t* __restrict__ xhi, uint16_t* __restrict__ xlo) {
    size_t i = ((size_t)blockIdx.x * 256 + threadIdx.x) * 4;
    f32x4 v = *(const f32x4*)(x + i);
    union { uint16_t s[4]; uint64_t q; } hi, lo;
#pragma unroll
    for (int j = 0; j < 4; ++j) {
        float f = v[j];
        uint16_t h = f32_to_bf16(f);
        hi.s[j] = h;
        lo.s[j] = f32_to_bf16(f - bf16_to_f32(h));
    }
    *(uint64_t*)(xhi + i) = hi.q;
    *(uint64_t*)(xlo + i) = lo.q;
}

// Binarize AND transpose W in one pass: wT[n][k] = (w_raw[k][n] > 0) as bf16.
// 64x64 tiles via LDS; coalesced f32x4 reads along N, coalesced 32B writes
// along K. Runs once per call; off the critical path.
__global__ void __launch_bounds__(256)
cvt_w_binary_T(const float* __restrict__ w_raw, uint16_t* __restrict__ wT) {
    __shared__ uint16_t sT[64 * 66];          // stride 66 halfs breaks bank symmetry
    const int t   = threadIdx.x;
    const int kt0 = blockIdx.y * 64;
    const int n0  = blockIdx.x * 64;

    // phase 1: read 16 floats per thread along N, binarize into LDS [k][n]
    {
        const int r  = t >> 2;                // 0..63  (k within tile)
        const int c0 = (t & 3) << 4;          // 0,16,32,48 (n within tile)
        const float* g = w_raw + (size_t)(kt0 + r) * NDIM + n0 + c0;
#pragma unroll
        for (int q = 0; q < 4; ++q) {
            f32x4 v = *(const f32x4*)(g + q * 4);
#pragma unroll
            for (int j = 0; j < 4; ++j)
                sT[r * 66 + c0 + q * 4 + j] = (v[j] > 0.0f) ? (uint16_t)0x3F80
                                                            : (uint16_t)0;
        }
    }
    __syncthreads();

    // phase 2: write wT[n][k], 32B contiguous per thread (coalesced)
    {
        const int n  = t >> 2;                // 0..63
        const int k0 = (t & 3) << 4;          // 0,16,32,48
        union { u32x4 q[2]; uint16_t s[16]; } o;
#pragma unroll
        for (int j = 0; j < 16; ++j) o.s[j] = sT[(k0 + j) * 66 + n];
        u32x4* d = (u32x4*)(wT + (size_t)(n0 + n) * KDIM + kt0 + k0);
        d[0] = o.q[0];
        d[1] = o.q[1];
    }
}

// Assemble a 16-half WMMA fragment from two 16B LDS chunks (ds_load_b128 x2).
struct B32B { u32x4 a, b; };
__device__ __forceinline__ v16bf load_frag2(const uint16_t* p0, const uint16_t* p1) {
    B32B t;
    t.a = *(const u32x4*)p0;
    t.b = *(const u32x4*)p1;
    return __builtin_bit_cast(v16bf, t);
}

__global__ void __launch_bounds__(256)
binary_dense_wmma(const uint16_t* __restrict__ xhi,
                  const uint16_t* __restrict__ xlo,
                  const uint16_t* __restrict__ wT,
                  float* __restrict__ out) {
    __shared__ __align__(16) uint16_t sAhi[BM * LDSA];
    __shared__ __align__(16) uint16_t sAlo[BM * LDSA];
    __shared__ __align__(16) uint16_t sBT [BN * LDSB];   // [n][k], pre-transposed

    const int t     = threadIdx.x;
    const int mBase = blockIdx.y * BM;
    const int nBase = blockIdx.x * BN;

    // 8 waves (wave32) in a 2x4 grid; each wave owns a 64x32 output region.
    const int wave = t >> 5;
    const int lane = t & 31;
    const int lr   = lane & 15;
    const int hiH  = lane >> 4;           // K-half selector per ISA layout
    const int wm   = (wave >> 2) * 64;
    const int wn   = (wave & 3) * 32;
    const int ak0  = hiH ? 8  : 0;        // 16-bit A 16x32: lanes>=16 hold K=8..15 / 24..31
    const int bk0  = hiH ? 8  : 0;        // 16-bit B 32x16: lanes>=16 hold K=8..15 / 24..31

    v8f zero;
#pragma unroll
    for (int e = 0; e < 8; ++e) zero[e] = 0.0f;
    v8f acc[4][2];
#pragma unroll
    for (int i = 0; i < 4; ++i)
#pragma unroll
        for (int j = 0; j < 2; ++j) acc[i][j] = zero;

    // Loader mapping (256 threads): identical pattern for A and B^T tiles.
    const int ar = t >> 1;                // tile row 0..127
    const int ac = (t & 1) << 4;          // halfs 0 or 16

    for (int kt = 0; kt < KDIM; kt += BK) {
        // ---- stage A hi/lo and B^T tiles (all b128 global + b128 LDS) ----
        {
            const u32x4* gh = (const u32x4*)(xhi + (size_t)(mBase + ar) * KDIM + kt + ac);
            const u32x4* gl = (const u32x4*)(xlo + (size_t)(mBase + ar) * KDIM + kt + ac);
            const u32x4* gb = (const u32x4*)(wT  + (size_t)(nBase + ar) * KDIM + kt + ac);
            u32x4 h0 = gh[0], h1 = gh[1];
            u32x4 l0 = gl[0], l1 = gl[1];
            u32x4 b0 = gb[0], b1 = gb[1];
            u32x4* dh = (u32x4*)(sAhi + ar * LDSA + ac);
            u32x4* dl = (u32x4*)(sAlo + ar * LDSA + ac);
            u32x4* db = (u32x4*)(sBT  + ar * LDSB + ac);
            dh[0] = h0; dh[1] = h1;
            dl[0] = l0; dl[1] = l1;
            db[0] = b0; db[1] = b1;
        }
        // ---- prefetch next K-tile (global_prefetch path; data is L2-resident) ----
        if (kt + BK < KDIM) {
            __builtin_prefetch(xhi + (size_t)(mBase + ar) * KDIM + kt + BK + ac, 0, 0);
            __builtin_prefetch(xlo + (size_t)(mBase + ar) * KDIM + kt + BK + ac, 0, 0);
            __builtin_prefetch(wT  + (size_t)(nBase + ar) * KDIM + kt + BK + ac, 0, 0);
        }
        __syncthreads();

        // ---- fragments (all contiguous ds_load_b128 pairs) ----
        v16bf bf[2];
#pragma unroll
        for (int j = 0; j < 2; ++j) {
            const uint16_t* p = sBT + (wn + j * 16 + lr) * LDSB + bk0;
            bf[j] = load_frag2(p, p + 16);              // K k0..k0+7 and k0+16..k0+23
        }
        v16bf ah[4], al[4];
#pragma unroll
        for (int i = 0; i < 4; ++i) {
            const uint16_t* ph = sAhi + (wm + i * 16 + lr) * LDSA + ak0;
            const uint16_t* pl = sAlo + (wm + i * 16 + lr) * LDSA + ak0;
            ah[i] = load_frag2(ph, ph + 16);            // K k0..k0+7 and k0+16..k0+23
            al[i] = load_frag2(pl, pl + 16);
        }

        // ---- 16 WMMAs per wave per K-step: hi and lo share one f32 accumulator ----
#pragma unroll
        for (int i = 0; i < 4; ++i)
#pragma unroll
            for (int j = 0; j < 2; ++j) {
                acc[i][j] = __builtin_amdgcn_wmma_f32_16x16x32_bf16(
                    false, ah[i], false, bf[j], (short)0, acc[i][j], false, false);
                acc[i][j] = __builtin_amdgcn_wmma_f32_16x16x32_bf16(
                    false, al[i], false, bf[j], (short)0, acc[i][j], false, false);
            }
        __syncthreads();
    }

    // ---- write out per ISA C/D layout: VGPR v -> M=v (lanes<16) / v+8 (lanes>=16) ----
#pragma unroll
    for (int i = 0; i < 4; ++i)
#pragma unroll
        for (int j = 0; j < 2; ++j) {
            int col = nBase + wn + j * 16 + lr;
#pragma unroll
            for (int v = 0; v < 8; ++v) {
                int row = mBase + wm + i * 16 + v + hiH * 8;
                out[(size_t)row * NDIM + col] = acc[i][j][v];
            }
        }
}

extern "C" void kernel_launch(void* const* d_in, const int* in_sizes, int n_in,
                              void* d_out, int out_size, void* d_ws, size_t ws_size,
                              hipStream_t stream) {
    (void)in_sizes; (void)n_in; (void)out_size; (void)ws_size;
    const float* x     = (const float*)d_in[0];
    const float* w_raw = (const float*)d_in[1];

    const size_t NEL = (size_t)MDIM * KDIM;        // 16,777,216 elems each
    uint16_t* xhi = (uint16_t*)d_ws;               // 32 MB
    uint16_t* xlo = xhi + NEL;                     // 32 MB
    uint16_t* wTb = xlo + NEL;                     // 32 MB (W^T, N x K)  -> 96 MB total

    const int cvtBlocksX = (int)(NEL / (256 * 4));
    cvt_x_split<<<cvtBlocksX, 256, 0, stream>>>(x, xhi, xlo);

    dim3 gridT(NDIM / 64, KDIM / 64);              // 64 x 64 transpose tiles
    cvt_w_binary_T<<<gridT, 256, 0, stream>>>(w_raw, wTb);

    dim3 grid(NDIM / BN, MDIM / BM);               // 32 x 32 = 1024 blocks, 8 waves each
    binary_dense_wmma<<<grid, 256, 0, stream>>>(xhi, xlo, wTb, (float*)d_out);
}
